// TransformerLayer_54941221650863
// MI455X (gfx1250) — matrix operand order repaired and checked by reading, not compiled, not run on hardware
//
#include <hip/hip_runtime.h>
#include <hip/hip_bf16.h>
#include <math.h>

// ---------------------------------------------------------------------------
// Problem constants (match reference)
// ---------------------------------------------------------------------------
#define NN   20000
#define NP   20032     // NN padded to a multiple of 64 (313 * 64) -> no guards
#define DD   256
#define KS   32        // sampled neighbors per node
#define NH   8
#define HD   32        // DD / NH
#define DFF  1024      // 4 * DD
#define EPS  1e-5f

typedef _Float16 v16h __attribute__((ext_vector_type(16)));
typedef _Float16 h8   __attribute__((ext_vector_type(8)));
typedef float    v8f  __attribute__((ext_vector_type(8)));

// ---------------------------------------------------------------------------
// Helpers
// ---------------------------------------------------------------------------
__device__ __forceinline__ float wave_sum32(float v) {
    #pragma unroll
    for (int off = 16; off > 0; off >>= 1) v += __shfl_xor(v, off, 32);
    return v;
}
__device__ __forceinline__ float wave_max32(float v) {
    #pragma unroll
    for (int off = 16; off > 0; off >>= 1) v = fmaxf(v, __shfl_xor(v, off, 32));
    return v;
}

// 32-bit LDS byte offset of a generic pointer that points into LDS
__device__ __forceinline__ unsigned lds_off(const void* p) {
    return (unsigned)(unsigned long long)(__attribute__((address_space(3))) const void*)p;
}

// CDNA5 async global -> LDS copy (16 bytes per lane), tracked by ASYNCcnt
__device__ __forceinline__ void async_copy_b128(unsigned lds_byte_off, const void* gptr) {
    unsigned long long ga = (unsigned long long)gptr;
    asm volatile("global_load_async_to_lds_b128 %0, %1, off"
                 :: "v"(lds_byte_off), "v"(ga) : "memory");
}
__device__ __forceinline__ void wait_async0() {
    asm volatile("s_wait_asynccnt 0" ::: "memory");
}

// ---------------------------------------------------------------------------
// Prep kernels: fp32 -> f16 convert, and convert+transpose weights
// WT[n*K + k] = W[k*Nout + n]   (row-major WT with rows of length K)
// ---------------------------------------------------------------------------
__global__ __launch_bounds__(256) void cvt_f16_kernel(const float* __restrict__ in,
                                                      _Float16* __restrict__ out, int n) {
    int i = blockIdx.x * 256 + threadIdx.x;
    if (i < n) out[i] = (_Float16)in[i];
}

__global__ __launch_bounds__(256) void transpose_f16_kernel(const float* __restrict__ W,
                                                            _Float16* __restrict__ WT,
                                                            int K, int Nout) {
    int i = blockIdx.x * 256 + threadIdx.x;
    if (i >= K * Nout) return;
    int n = i / K;
    int k = i - n * K;
    WT[i] = (_Float16)W[(size_t)k * Nout + n];
}

// ---------------------------------------------------------------------------
// WMMA GEMM:  Out[NP,Nout] = A[NP,Kdim] (f16) @ BT[Nout,Kdim]^T (f16) + bias
// Block: 256 threads = 8 waves; block tile 64(M) x 256(N); BK = 32.
// Each wave computes a 32x64 sub-tile = 2x4 v_wmma_f32_16x16x32_f16,
// reusing each A fragment across 4 B tiles (2x WMMA per LDS byte vs 32x32).
// Tiles staged with CDNA5 async global->LDS copies (ASYNCcnt).
// All buffers are padded to NP rows -> no bounds checks anywhere.
// ---------------------------------------------------------------------------
template <bool RELU, bool OUTF32>
__global__ __launch_bounds__(256) void gemm_wmma_kernel(const _Float16* __restrict__ A,
                                                        const _Float16* __restrict__ BT,
                                                        const float* __restrict__ bias,
                                                        void* __restrict__ Out,
                                                        int Kdim, int Nout) {
    __shared__ _Float16 sA[64 * 40];    // 64 rows, stride 40 (pad vs bank conflicts)
    __shared__ _Float16 sB[256 * 40];   // 256 output cols, stride 40

    const int tid  = threadIdx.x;
    const int lane = tid & 31;
    const int w    = tid >> 5;
    const int m0   = blockIdx.x * 64;
    const int n0   = blockIdx.y * 256;

    const int wm = (w >> 2) * 32;       // wave M offset inside block tile
    const int wn = (w & 3) * 64;        // wave N offset inside block tile
    const int lr = lane & 15;           // lane index inside 16-wide tile
    const int lh = lane >> 4;           // lane half (0/1)

    v8f acc[2][4] = {};

    // staging assignments (per thread, all unguarded thanks to NP padding)
    const int ar = tid >> 2, ac = (tid & 3) * 8;   // A: 8 f16 (16B) per thread
    const unsigned sAoff = lds_off(sA + ar * 40 + ac);
    const unsigned sBoff = lds_off(sB + tid * 40);          // B: full 32-f16 row
    const _Float16* Abase = A  + (size_t)(m0 + ar) * Kdim + ac;
    const _Float16* Bbase = BT + (size_t)(n0 + tid) * Kdim;

    for (int k0 = 0; k0 < Kdim; k0 += 32) {
        // ---- async stage A tile (64x32) and B tile (256x32) into LDS ----
        async_copy_b128(sAoff, Abase + k0);
        #pragma unroll
        for (int c = 0; c < 4; ++c)
            async_copy_b128(sBoff + c * 16, Bbase + k0 + c * 8);
        if (k0 + 32 < Kdim) {
            __builtin_prefetch(Abase + k0 + 32, 0, 3);
            __builtin_prefetch(Bbase + k0 + 32, 0, 3);
        }
        wait_async0();
        __syncthreads();

        // ---- build fragments per documented 16-bit A/B VGPR layouts ----
        v16h afrag[2], bfrag[4];
        #pragma unroll
        for (int t = 0; t < 2; ++t) {
            // A 16x32: lane row = lr; element j -> K = (j/8)*16 + lh*8 + (j%8)
            const int arow = wm + t * 16 + lr;
            h8 a0 = *(const h8*)(sA + arow * 40 + lh * 8);
            h8 a1 = *(const h8*)(sA + arow * 40 + 16 + lh * 8);
            v16h av;
            #pragma unroll
            for (int j = 0; j < 8; ++j) { av[j] = a0[j]; av[8 + j] = a1[j]; }
            afrag[t] = av;
        }
        #pragma unroll
        for (int t = 0; t < 4; ++t) {
            // B 32x16: lane col = lr; element j -> K = (j/8)*16 + lh*8 + (j%8)
            const int bcol = wn + t * 16 + lr;
            h8 b0 = *(const h8*)(sB + bcol * 40 + lh * 8);
            h8 b1 = *(const h8*)(sB + bcol * 40 + 16 + lh * 8);
            v16h bv;
            #pragma unroll
            for (int j = 0; j < 8; ++j) { bv[j] = b0[j]; bv[8 + j] = b1[j]; }
            bfrag[t] = bv;
        }

        #pragma unroll
        for (int mt = 0; mt < 2; ++mt)
            #pragma unroll
            for (int nt = 0; nt < 4; ++nt)
                acc[mt][nt] = __builtin_amdgcn_wmma_f32_16x16x32_f16(
                    false, afrag[mt], false, bfrag[nt],
                    (short)0, acc[mt][nt], false, false);

        __syncthreads();
    }

    // ---- epilogue: bias (+ReLU), store f32 or f16 (unguarded: NP-padded) ----
    #pragma unroll
    for (int mt = 0; mt < 2; ++mt) {
        #pragma unroll
        for (int nt = 0; nt < 4; ++nt) {
            const int col = n0 + wn + nt * 16 + lr;
            const float bcolv = bias[col];
            #pragma unroll
            for (int r = 0; r < 8; ++r) {
                // C/D layout: VGPR r -> M = r + 8*lane_half ; N = lane % 16
                const int row = m0 + wm + mt * 16 + r + 8 * lh;
                float v = acc[mt][nt][r] + bcolv;
                if (RELU) v = fmaxf(v, 0.0f);
                if (OUTF32)
                    ((float*)Out)[(size_t)row * Nout + col] = v;
                else
                    ((_Float16*)Out)[(size_t)row * Nout + col] = (_Float16)v;
            }
        }
    }
}

// ---------------------------------------------------------------------------
// Attention kernel: 1 wave (32 lanes) per node, 4 nodes per 128-thread block.
// scores pass: lane = key index; softmax via wave shuffles;
// output pass: lane = head dim; weights broadcast via shuffles.
// Q/K/V tables are ~10 MB f16 each -> resident in the 192 MB L2.
// ---------------------------------------------------------------------------
__global__ __launch_bounds__(128) void attn_kernel(const _Float16* __restrict__ Q,
                                                   const _Float16* __restrict__ Kp,
                                                   const _Float16* __restrict__ Vp,
                                                   const int* __restrict__ samples,
                                                   _Float16* __restrict__ O) {
    const int lane = threadIdx.x & 31;
    const int wave = threadIdx.x >> 5;
    const int node = blockIdx.x * 4 + wave;
    if (node >= NN) return;

    const float scale = 0.17677669529663687f;        // 1/sqrt(32)
    const int   idx   = samples[node * KS + lane];   // this lane's key row

    #pragma unroll
    for (int h = 0; h < NH; ++h) {
        const size_t qb = (size_t)node * DD + h * HD;
        const size_t kb = (size_t)idx  * DD + h * HD;

        // ---- scores: lane = key ----
        float s = 0.0f;
        #pragma unroll
        for (int c = 0; c < 4; ++c) {
            h8 qv = *(const h8*)(Q  + qb + c * 8);
            h8 kv = *(const h8*)(Kp + kb + c * 8);
            #pragma unroll
            for (int j = 0; j < 8; ++j) s += (float)qv[j] * (float)kv[j];
        }
        s *= scale;

        // ---- softmax over 32 keys (entire wave) ----
        const float m   = wave_max32(s);
        const float e   = expf(s - m);
        const float sm  = wave_sum32(e);
        const float wgt = e / sm;

        // ---- output: lane = head dim ----
        float o = 0.0f;
        #pragma unroll
        for (int k = 0; k < KS; ++k) {
            const float wk = __shfl(wgt, k, 32);
            const int   ik = __shfl(idx, k, 32);
            o += wk * (float)Vp[(size_t)ik * DD + h * HD + lane];
        }
        O[(size_t)node * DD + h * HD + lane] = (_Float16)o;
    }
}

// ---------------------------------------------------------------------------
// Residual + LayerNorm: 1 wave per row (D=256 -> 8 elems/lane), 8 rows/block.
// out_f32 always written; out_f16 optionally (next GEMM's A operand).
// ---------------------------------------------------------------------------
__global__ __launch_bounds__(256) void res_ln_kernel(const float* __restrict__ Y,
                                                     const float* __restrict__ R,
                                                     const float* __restrict__ gamma,
                                                     const float* __restrict__ beta,
                                                     float* __restrict__ out_f32,
                                                     _Float16* __restrict__ out_f16,
                                                     int rows) {
    const int lane = threadIdx.x & 31;
    const int wave = threadIdx.x >> 5;
    const int row  = blockIdx.x * 8 + wave;
    if (row >= rows) return;

    float v[8];
    float s = 0.0f;
    #pragma unroll
    for (int j = 0; j < 8; ++j) {
        const int c = j * 32 + lane;
        v[j] = Y[(size_t)row * DD + c] + R[(size_t)row * DD + c];
        s += v[j];
    }
    const float mu = wave_sum32(s) * (1.0f / DD);

    float q = 0.0f;
    #pragma unroll
    for (int j = 0; j < 8; ++j) { const float d = v[j] - mu; q += d * d; }
    const float var  = wave_sum32(q) * (1.0f / DD);
    const float rstd = rsqrtf(var + EPS);

    #pragma unroll
    for (int j = 0; j < 8; ++j) {
        const int c = j * 32 + lane;
        const float o = (v[j] - mu) * rstd * gamma[c] + beta[c];
        out_f32[(size_t)row * DD + c] = o;
        if (out_f16) out_f16[(size_t)row * DD + c] = (_Float16)o;
    }
}

// ---------------------------------------------------------------------------
// Host-side launcher
// ---------------------------------------------------------------------------
extern "C" void kernel_launch(void* const* d_in, const int* in_sizes, int n_in,
                              void* d_out, int out_size, void* d_ws, size_t ws_size,
                              hipStream_t stream) {
    const float* x   = (const float*)d_in[0];
    const int*   smp = (const int*)  d_in[1];
    const float* Wq  = (const float*)d_in[2];  const float* bq = (const float*)d_in[3];
    const float* Wk  = (const float*)d_in[4];  const float* bk = (const float*)d_in[5];
    const float* Wv  = (const float*)d_in[6];  const float* bv = (const float*)d_in[7];
    const float* Wo  = (const float*)d_in[8];  const float* bo = (const float*)d_in[9];
    const float* W1  = (const float*)d_in[10]; const float* b1 = (const float*)d_in[11];
    const float* W2  = (const float*)d_in[12]; const float* b2 = (const float*)d_in[13];
    const float* g1  = (const float*)d_in[14]; const float* be1= (const float*)d_in[15];
    const float* g2  = (const float*)d_in[16]; const float* be2= (const float*)d_in[17];
    float* out = (float*)d_out;

    // ---- workspace carving (~104 MB), all N-row buffers padded to NP rows ----
    char* ws = (char*)d_ws;
    const size_t ndh = (size_t)NP * DD * sizeof(_Float16);
    const size_t ndf = (size_t)NP * DD * sizeof(float);
    size_t off = 0;
    _Float16* xh  = (_Float16*)(ws + off); off += ndh;
    _Float16* Qh  = (_Float16*)(ws + off); off += ndh;   // Q/K/V/O region ==
    _Float16* Kh  = (_Float16*)(ws + off); off += ndh;   //   NP*DFF f16 total,
    _Float16* Vh  = (_Float16*)(ws + off); off += ndh;   //   aliased by Hf
    _Float16* Oh  = (_Float16*)(ws + off); off += ndh;   //   after attention
    _Float16* Hf  = Qh;                                  // NP x DFF f16 alias
    float*    Yb  = (float*)(ws + off); off += ndf;      // reused proj output
    float*    x1f = (float*)(ws + off); off += ndf;
    _Float16* x1h = (_Float16*)(ws + off); off += ndh;
    _Float16* WqT = (_Float16*)(ws + off); off += (size_t)DD * DD * 2;
    _Float16* WkT = (_Float16*)(ws + off); off += (size_t)DD * DD * 2;
    _Float16* WvT = (_Float16*)(ws + off); off += (size_t)DD * DD * 2;
    _Float16* WoT = (_Float16*)(ws + off); off += (size_t)DD * DD * 2;
    _Float16* W1T = (_Float16*)(ws + off); off += (size_t)DD * DFF * 2;
    _Float16* W2T = (_Float16*)(ws + off); off += (size_t)DFF * DD * 2;
    (void)ws_size; (void)in_sizes; (void)n_in; (void)out_size;

    // ---- prep: f16 activations + transposed f16 weights ----
    cvt_f16_kernel<<<((size_t)NN * DD + 255) / 256, 256, 0, stream>>>(x, xh, NN * DD);
    transpose_f16_kernel<<<(DD * DD + 255) / 256, 256, 0, stream>>>(Wq, WqT, DD, DD);
    transpose_f16_kernel<<<(DD * DD + 255) / 256, 256, 0, stream>>>(Wk, WkT, DD, DD);
    transpose_f16_kernel<<<(DD * DD + 255) / 256, 256, 0, stream>>>(Wv, WvT, DD, DD);
    transpose_f16_kernel<<<(DD * DD + 255) / 256, 256, 0, stream>>>(Wo, WoT, DD, DD);
    transpose_f16_kernel<<<(DD * DFF + 255) / 256, 256, 0, stream>>>(W1, W1T, DD, DFF);
    transpose_f16_kernel<<<(DFF * DD + 255) / 256, 256, 0, stream>>>(W2, W2T, DFF, DD);

    const int mblk = NP / 64;   // 313

    // ---- Q/K/V projections (project ALL rows once; gather projected K/V) ----
    gemm_wmma_kernel<false, false><<<dim3(mblk, DD / 256), 256, 0, stream>>>(
        xh, WqT, bq, Qh, DD, DD);
    gemm_wmma_kernel<false, false><<<dim3(mblk, DD / 256), 256, 0, stream>>>(
        xh, WkT, bk, Kh, DD, DD);
    gemm_wmma_kernel<false, false><<<dim3(mblk, DD / 256), 256, 0, stream>>>(
        xh, WvT, bv, Vh, DD, DD);

    // ---- sampled-neighbor attention (L2-resident gather) ----
    attn_kernel<<<(NN + 3) / 4, 128, 0, stream>>>(Qh, Kh, Vh, smp, Oh);

    // ---- out projection + residual LN 1 ----
    gemm_wmma_kernel<false, true><<<dim3(mblk, DD / 256), 256, 0, stream>>>(
        Oh, WoT, bo, Yb, DD, DD);
    res_ln_kernel<<<(NN + 7) / 8, 256, 0, stream>>>(Yb, x, g1, be1, x1f, x1h, NN);

    // ---- FFN: relu(x1 @ W1 + b1) @ W2 + b2, residual LN 2 ----
    gemm_wmma_kernel<true, false><<<dim3(mblk, DFF / 256), 256, 0, stream>>>(
        x1h, W1T, b1, Hf, DD, DFF);
    gemm_wmma_kernel<false, true><<<dim3(mblk, DD / 256), 256, 0, stream>>>(
        Hf, W2T, b2, Yb, DFF, DD);
    res_ln_kernel<<<(NN + 7) / 8, 256, 0, stream>>>(Yb, x1f, g2, be2, out, nullptr, NN);
}
